// WindowAttention_36893769072672
// MI455X (gfx1250) — compile-verified, matrix operand order checked by reading, not run on hardware
//
#include <hip/hip_runtime.h>
#include <hip/hip_bf16.h>

typedef __attribute__((ext_vector_type(16))) __bf16 v16bf;
typedef __attribute__((ext_vector_type(8)))  __bf16 v8bf;
typedef __attribute__((ext_vector_type(8)))  float  v8f;

#define DDIM 384
#define NDIM 1152
#define HD   32
#define NH   12
#define TT   49
#define EPSV 1e-6f
#define LDP  72     // padded LDS row stride (bf16 elems) to break bank conflicts

#define WMMA_BF16(a, b, c) \
    __builtin_amdgcn_wmma_f32_16x16x32_bf16(false, (a), false, (b), (short)0, (c), false, false)

// 16-lane-half XOR reductions via ds_swizzle (group-of-32 mode, and_mask=0x1f)
#define SWIZ_XOR(x, m) __builtin_amdgcn_ds_swizzle((x), (0x1f | ((m) << 10)))

static __device__ __forceinline__ float red16_add(float v) {
    v += __int_as_float(SWIZ_XOR(__float_as_int(v), 1));
    v += __int_as_float(SWIZ_XOR(__float_as_int(v), 2));
    v += __int_as_float(SWIZ_XOR(__float_as_int(v), 4));
    v += __int_as_float(SWIZ_XOR(__float_as_int(v), 8));
    return v;
}
static __device__ __forceinline__ float red16_max(float v) {
    v = fmaxf(v, __int_as_float(SWIZ_XOR(__float_as_int(v), 1)));
    v = fmaxf(v, __int_as_float(SWIZ_XOR(__float_as_int(v), 2)));
    v = fmaxf(v, __int_as_float(SWIZ_XOR(__float_as_int(v), 4)));
    v = fmaxf(v, __int_as_float(SWIZ_XOR(__float_as_int(v), 8)));
    return v;
}

union FragU { v16bf v; v8bf h[2]; };

// Lane's 16 bf16: two contiguous 8-elem chunks at K = khalf*8 and 16+khalf*8
// (ISA 16-bit A/B layout, wave32). Both chunks are single b128 loads.
static __device__ __forceinline__ v16bf frag_from_row(const __bf16* row, int khalf) {
    FragU f;
    f.h[0] = *(const v8bf*)(row + khalf * 8);
    f.h[1] = *(const v8bf*)(row + 16 + khalf * 8);
    return f.v;
}

static __device__ __forceinline__ v16bf frag_zero() {
    FragU f;
#pragma unroll
    for (int i = 0; i < 16; ++i) f.v[i] = (__bf16)0.0f;
    return f.v;
}

// CDNA5 async direct-to-LDS copy (16B per lane), tracked by ASYNCcnt.
static __device__ __forceinline__ void async_copy_b128(unsigned lds_addr, const void* gaddr) {
    asm volatile("global_load_async_to_lds_b128 %0, %1, off"
                 :: "v"(lds_addr), "v"((unsigned long long)(uintptr_t)gaddr)
                 : "memory");
}
static __device__ __forceinline__ void wait_async0() {
    asm volatile("s_wait_asynccnt 0" ::: "memory");
}

// ---------------------------------------------------------------------------
// Kernel 1: qkv = x @ qkv_w.T (M=200704,N=1152,K=384), fused RMS-norm, q scaled
// by hd^-0.5, bf16 out. 256 thr / 8 waves; tile 128x64; BK=64, double-buffered
// LDS (one barrier per K-chunk, global loads overlap WMMA).
// Wave's 32 N-columns are exactly one head -> in-register RMS reduction.
// ---------------------------------------------------------------------------
__global__ __launch_bounds__(256) void qkv_kernel(
    const float* __restrict__ x, const float* __restrict__ qkv_w,
    const float* __restrict__ qnw, const float* __restrict__ knw,
    __bf16* __restrict__ qkv_out)
{
    __shared__ __bf16 As[2][128][LDP];
    __shared__ __bf16 Ws[2][64][LDP];

    const int tid  = threadIdx.x;
    const int lane = tid & 31;
    const int wave = tid >> 5;
    const int wm   = wave >> 1;          // 0..3
    const int wn   = wave & 1;           // 0..1
    const int bm   = blockIdx.x * 128;
    const int bn   = blockIdx.y * 64;
    const int lrow  = lane & 15;
    const int khalf = lane >> 4;

    v8f acc[2][2];
    v8f zero = {};
#pragma unroll
    for (int i = 0; i < 2; ++i)
#pragma unroll
        for (int j = 0; j < 2; ++j) acc[i][j] = zero;

    auto stage = [&](int kc, int buf) {
        for (int i = tid; i < 128 * 64; i += 256) {
            int r = i >> 6, c = i & 63;
            As[buf][r][c] = (__bf16)x[(size_t)(bm + r) * DDIM + kc + c];
        }
        for (int i = tid; i < 64 * 64; i += 256) {
            int r = i >> 6, c = i & 63;
            Ws[buf][r][c] = (__bf16)qkv_w[(size_t)(bn + r) * DDIM + kc + c];
        }
    };

    stage(0, 0);
    __syncthreads();

    for (int it = 0; it < DDIM / 64; ++it) {
        if (it < DDIM / 64 - 1) stage((it + 1) * 64, (it + 1) & 1);
        const int buf = it & 1;
#pragma unroll
        for (int s = 0; s < 2; ++s) {
            v16bf a[2], b[2];
#pragma unroll
            for (int t = 0; t < 2; ++t)
                a[t] = frag_from_row(&As[buf][wm * 32 + t * 16 + lrow][s * 32], khalf);
#pragma unroll
            for (int t = 0; t < 2; ++t)
                b[t] = frag_from_row(&Ws[buf][wn * 32 + t * 16 + lrow][s * 32], khalf);
#pragma unroll
            for (int ti = 0; ti < 2; ++ti)
#pragma unroll
                for (int tj = 0; tj < 2; ++tj)
                    acc[ti][tj] = WMMA_BF16(a[ti], b[tj], acc[ti][tj]);
        }
        __syncthreads();
    }

    // Epilogue: RMS-norm per (row, head) over this wave's 32 columns.
    const int ncol0 = bn + wn * 32;       // head-aligned
    const int sect  = ncol0 / DDIM;       // 0=q, 1=k, 2=v
    const float qscale = rsqrtf((float)HD);
    float nw0 = 1.0f, nw1 = 1.0f;         // hoisted, q-scale folded in
    if (sect == 0) { nw0 = qnw[lrow] * qscale; nw1 = qnw[16 + lrow] * qscale; }
    else if (sect == 1) { nw0 = knw[lrow]; nw1 = knw[16 + lrow]; }

#pragma unroll
    for (int ti = 0; ti < 2; ++ti) {
#pragma unroll
        for (int r = 0; r < 8; ++r) {
            int row = bm + wm * 32 + ti * 16 + r + khalf * 8;
            float v0 = acc[ti][0][r], v1 = acc[ti][1][r];
            if (sect < 2) {
                float ss = red16_add(v0 * v0 + v1 * v1);
                float rms = rsqrtf(ss * (1.0f / 32.0f) + EPSV);
                v0 = v0 * rms * nw0;
                v1 = v1 * rms * nw1;
            }
            size_t base = (size_t)row * NDIM + ncol0;
            qkv_out[base + lrow]      = (__bf16)v0;
            qkv_out[base + 16 + lrow] = (__bf16)v1;
        }
    }
}

// ---------------------------------------------------------------------------
// Kernel 2: per-window attention. One block per window, one wave per head.
// Block-level staging of V transposed (VT[d][token], padded) so P@V WMMA
// operands come from LDS as b128 fragments. Softmax fully in registers with
// ds_swizzle reductions; P relayout through per-wave LDS buffer.
// Dynamic LDS: VT (12*32*72) + 8 waves * (64*72) bf16.
// ---------------------------------------------------------------------------
__global__ __launch_bounds__(256) void attn_kernel(
    const __bf16* __restrict__ qkv,
    const float* __restrict__ bias_table,
    const int* __restrict__ rel_index,
    __bf16* __restrict__ attn_out)
{
    extern __shared__ __bf16 smem[];
    __bf16* VT = smem;                               // [DDIM][LDP] transposed V
    __bf16* P  = smem + DDIM * LDP + (threadIdx.x >> 5) * (64 * LDP);

    const int b     = blockIdx.x;
    const int lane  = threadIdx.x & 31;
    const int wave  = threadIdx.x >> 5;
    const int lrow  = lane & 15;
    const int khalf = lane >> 4;
    const size_t tokBase = (size_t)b * TT;

    // ---- stage V transposed: VT[c][t] = v[t][c], c = h*32+d, coalesced ----
    for (int t = wave; t < TT; t += 8)
        for (int c = lane; c < DDIM; c += 32)
            VT[c * LDP + t] = qkv[(tokBase + t) * NDIM + 2 * DDIM + c];
    for (int t = TT + wave; t < 64; t += 8)
        for (int c = lane; c < DDIM; c += 32)
            VT[c * LDP + t] = (__bf16)0.0f;
    __syncthreads();

    for (int h = wave; h < NH; h += 8) {
        // ---- S = q @ k^T : 4x4 tiles, K = hd = 32 (single WMMA step) ----
        v16bf qf[4], kf[4];
#pragma unroll
        for (int t = 0; t < 4; ++t) {
            int m = t * 16 + lrow;
            qf[t] = (m < TT) ? frag_from_row(qkv + (tokBase + m) * NDIM + h * HD, khalf)
                             : frag_zero();
            kf[t] = (m < TT) ? frag_from_row(qkv + (tokBase + m) * NDIM + DDIM + h * HD, khalf)
                             : frag_zero();
        }
        v8f S[4][4];
        v8f zero = {};
#pragma unroll
        for (int ti = 0; ti < 4; ++ti)
#pragma unroll
            for (int tj = 0; tj < 4; ++tj)
                S[ti][tj] = WMMA_BF16(qf[ti], kf[tj], zero);

        // ---- bias + masked softmax ----
#pragma unroll
        for (int ti = 0; ti < 4; ++ti) {
#pragma unroll
            for (int r = 0; r < 8; ++r) {
                int q = ti * 16 + r + khalf * 8;
                float vv[4];
#pragma unroll
                for (int tj = 0; tj < 4; ++tj) {
                    int kk = tj * 16 + lrow;
                    float s = S[ti][tj][r];
                    if (kk < TT && q < TT)
                        s += bias_table[rel_index[q * TT + kk] * NH + h];
                    if (kk >= TT) s = -1e30f;
                    vv[tj] = s;
                }
                float mx = red16_max(fmaxf(fmaxf(vv[0], vv[1]), fmaxf(vv[2], vv[3])));
                float sum = 0.f;
#pragma unroll
                for (int tj = 0; tj < 4; ++tj) { vv[tj] = __expf(vv[tj] - mx); sum += vv[tj]; }
                sum = red16_add(sum);
                float inv = 1.0f / sum;
#pragma unroll
                for (int tj = 0; tj < 4; ++tj)
                    P[q * LDP + tj * 16 + lrow] = (__bf16)(vv[tj] * inv);
            }
        }

        // ---- O = P @ V : M=64(4), N=32(2), K=64(2 chunks), all LDS b128 ----
        v8f O[4][2];
#pragma unroll
        for (int ti = 0; ti < 4; ++ti)
#pragma unroll
            for (int tn = 0; tn < 2; ++tn) O[ti][tn] = zero;

#pragma unroll
        for (int c = 0; c < 2; ++c) {
            v16bf pf[4], vf[2];
#pragma unroll
            for (int ti = 0; ti < 4; ++ti)
                pf[ti] = frag_from_row(&P[(ti * 16 + lrow) * LDP + c * 32], khalf);
#pragma unroll
            for (int tn = 0; tn < 2; ++tn)
                vf[tn] = frag_from_row(&VT[(h * HD + tn * 16 + lrow) * LDP + c * 32], khalf);
#pragma unroll
            for (int ti = 0; ti < 4; ++ti)
#pragma unroll
                for (int tn = 0; tn < 2; ++tn)
                    O[ti][tn] = WMMA_BF16(pf[ti], vf[tn], O[ti][tn]);
        }

        // ---- store rows < 49, bf16 intermediate ----
#pragma unroll
        for (int ti = 0; ti < 4; ++ti) {
#pragma unroll
            for (int r = 0; r < 8; ++r) {
                int q = ti * 16 + r + khalf * 8;
                if (q < TT) {
                    size_t o = (tokBase + q) * DDIM + h * HD;
                    attn_out[o + lrow]      = (__bf16)O[ti][0][r];
                    attn_out[o + 16 + lrow] = (__bf16)O[ti][1][r];
                }
            }
        }
    }
}

// ---------------------------------------------------------------------------
// Kernel 3: out = attn @ proj_w.T + proj_b (M=200704,N=384,K=384), fp32 out.
// A operand staged by async direct-to-LDS b128 copies (ASYNCcnt path),
// double-buffered so async copies for chunk i+1 overlap WMMA on chunk i.
// ---------------------------------------------------------------------------
__global__ __launch_bounds__(256) void proj_kernel(
    const __bf16* __restrict__ a, const float* __restrict__ w,
    const float* __restrict__ bias, float* __restrict__ out)
{
    __shared__ __bf16 As[2][128][LDP];
    __shared__ __bf16 Ws[2][64][LDP];

    const int tid  = threadIdx.x;
    const int lane = tid & 31;
    const int wave = tid >> 5;
    const int wm   = wave >> 1;
    const int wn   = wave & 1;
    const int bm   = blockIdx.x * 128;
    const int bn   = blockIdx.y * 64;
    const int lrow  = lane & 15;
    const int khalf = lane >> 4;
    const unsigned lbase  = (unsigned)(uintptr_t)&As[0][0][0];
    const unsigned abufsz = (unsigned)(128 * LDP * sizeof(__bf16));

    v8f acc[2][2];
    v8f zero = {};
#pragma unroll
    for (int i = 0; i < 2; ++i)
#pragma unroll
        for (int j = 0; j < 2; ++j) acc[i][j] = zero;

    auto stage = [&](int kc, int buf) {
        // A tile: 128x64 bf16 = 1024 x 16B chunks, async global->LDS
#pragma unroll
        for (int j = 0; j < 4; ++j) {
            int ci = tid + j * 256;
            int r  = ci >> 3, c8 = ci & 7;
            async_copy_b128(lbase + (unsigned)buf * abufsz + (unsigned)(r * LDP + c8 * 8) * 2,
                            a + (size_t)(bm + r) * DDIM + kc + c8 * 8);
        }
        // W tile: fp32 -> bf16 through VGPRs
        for (int i = tid; i < 64 * 64; i += 256) {
            int r = i >> 6, c = i & 63;
            Ws[buf][r][c] = (__bf16)w[(size_t)(bn + r) * DDIM + kc + c];
        }
    };

    stage(0, 0);
    wait_async0();
    __syncthreads();

    for (int it = 0; it < DDIM / 64; ++it) {
        const bool more = (it < DDIM / 64 - 1);
        if (more) stage((it + 1) * 64, (it + 1) & 1);
        const int buf = it & 1;
#pragma unroll
        for (int s = 0; s < 2; ++s) {
            v16bf af[2], bf[2];
#pragma unroll
            for (int t = 0; t < 2; ++t)
                af[t] = frag_from_row(&As[buf][wm * 32 + t * 16 + lrow][s * 32], khalf);
#pragma unroll
            for (int t = 0; t < 2; ++t)
                bf[t] = frag_from_row(&Ws[buf][wn * 32 + t * 16 + lrow][s * 32], khalf);
#pragma unroll
            for (int ti = 0; ti < 2; ++ti)
#pragma unroll
                for (int tj = 0; tj < 2; ++tj)
                    acc[ti][tj] = WMMA_BF16(af[ti], bf[tj], acc[ti][tj]);
        }
        if (more) wait_async0();
        __syncthreads();
    }

    const int ncol0 = bn + wn * 32;
    const float b0 = bias[ncol0 + lrow];
    const float b1 = bias[ncol0 + 16 + lrow];
#pragma unroll
    for (int ti = 0; ti < 2; ++ti) {
#pragma unroll
        for (int r = 0; r < 8; ++r) {
            int row = bm + wm * 32 + ti * 16 + r + khalf * 8;
            size_t base = (size_t)row * DDIM + ncol0;
            out[base + lrow]      = acc[ti][0][r] + b0;
            out[base + 16 + lrow] = acc[ti][1][r] + b1;
        }
    }
}

extern "C" void kernel_launch(void* const* d_in, const int* in_sizes, int n_in,
                              void* d_out, int out_size, void* d_ws, size_t ws_size,
                              hipStream_t stream) {
    const float* x          = (const float*)d_in[0];
    const float* qkv_w      = (const float*)d_in[1];
    const float* q_norm_w   = (const float*)d_in[2];
    const float* k_norm_w   = (const float*)d_in[3];
    const float* proj_w     = (const float*)d_in[4];
    const float* proj_b     = (const float*)d_in[5];
    const float* bias_table = (const float*)d_in[6];
    const int*   rel_index  = (const int*)d_in[7];
    float* out = (float*)d_out;

    const int M = 4096 * TT;                   // 200704, divisible by 128
    __bf16* qkv  = (__bf16*)d_ws;              // M x 1152 bf16
    __bf16* attn = qkv + (size_t)M * NDIM;     // M x 384 bf16

    qkv_kernel<<<dim3(M / 128, NDIM / 64), 256, 0, stream>>>(
        x, qkv_w, q_norm_w, k_norm_w, qkv);

    const size_t attn_smem = (size_t)(DDIM * LDP + 8 * 64 * LDP) * sizeof(__bf16);
    attn_kernel<<<dim3(4096), 256, attn_smem, stream>>>(
        qkv, bias_table, rel_index, attn);

    proj_kernel<<<dim3(M / 128, DDIM / 64), 256, 0, stream>>>(
        attn, proj_w, proj_b, out);
}
